// FactorizedJumpOperator_89215060673158
// MI455X (gfx1250) — matrix operand order repaired and checked by reading, not compiled, not run on hardware
//
#include <hip/hip_runtime.h>
#include <hip/hip_bf16.h>

// ---------------------------------------------------------------------------
// FactorizedJumpOperator on gfx1250 (MI455X, wave32, WMMA).
//
// Pipeline (all on `stream`, graph-capture safe):
//   1) k_init    : zero bucket counters, fill sorted-index array with -1,
//                  split f32 weights into bf16 hi/lo pairs in workspace.
//   2) k_count   : histogram rows into 256 (src,tgt) buckets (atomics).
//   3) k_scan    : single block: pad buckets to multiples of 32 rows,
//                  exclusive scan, build tile->bucket map, publish tile count.
//   4) k_scatter : scatter row indices into bucket-sorted order (atomics).
//   5) k_jump    : 1 wave per 32-row tile (2 M-subtiles share each B fragment
//                  to halve L2 weight traffic); bf16x2-split WMMA GEMMs
//                  (16x16x32 bf16, f32 accum) for encode and decode.
// ---------------------------------------------------------------------------

#define NCH   16
#define DIM   128
#define BATCH 131072
#define NPAIR 256
#define ROWS  32                              // rows per wave/tile
#define NTILE_MAX (BATCH / ROWS + NPAIR)      // 4352 worst-case padded tiles

typedef __attribute__((ext_vector_type(16))) __bf16 v16bf;
typedef __attribute__((ext_vector_type(8)))  float  v8f;
typedef __attribute__((ext_vector_type(4)))  float  v4f;

union FragB16 { v16bf bf; unsigned int u[8]; };

// ---- workspace layout (ints) ----------------------------------------------
#define WS_COUNTS   0
#define WS_OFFSETS  256
#define WS_CURSORS  512
#define WS_NTILES   768
#define WS_TILEB    1024
#define WS_SORTED   (WS_TILEB + NTILE_MAX)                 // 5376
#define WS_WEIGHTS  (WS_SORTED + NTILE_MAX * ROWS)         // 144640 (int units)
#define WELEMS      (NCH * DIM * DIM)                      // 262144 bf16 each

__device__ __forceinline__ unsigned short f32_to_bf16_rne(float f) {
    unsigned int x = __builtin_bit_cast(unsigned int, f);
    unsigned int r = x + 0x7FFFu + ((x >> 16) & 1u);
    return (unsigned short)(r >> 16);
}
__device__ __forceinline__ float bf16_bits_to_f32(unsigned short h) {
    unsigned int x = ((unsigned int)h) << 16;
    return __builtin_bit_cast(float, x);
}
__device__ __forceinline__ void split_bf16(float f, unsigned short& hi, unsigned short& lo) {
    hi = f32_to_bf16_rne(f);
    lo = f32_to_bf16_rne(f - bf16_bits_to_f32(hi));
}

// ---------------------------------------------------------------------------
__global__ void k_init(const float* __restrict__ enc_w, const float* __restrict__ dec_w,
                       int* __restrict__ counts, int* __restrict__ sorted,
                       unsigned short* __restrict__ ehi, unsigned short* __restrict__ elo,
                       unsigned short* __restrict__ dhi, unsigned short* __restrict__ dlo) {
    int i = blockIdx.x * blockDim.x + threadIdx.x;
    if (i < NPAIR) counts[i] = 0;
    if (i < NTILE_MAX * ROWS) sorted[i] = -1;
    if (i < WELEMS) {
        unsigned short h, l;
        split_bf16(enc_w[i], h, l);  ehi[i] = h;  elo[i] = l;
        split_bf16(dec_w[i], h, l);  dhi[i] = h;  dlo[i] = l;
    }
}

__global__ void k_count(const int* __restrict__ src, const int* __restrict__ tgt,
                        int* __restrict__ counts) {
    int i = blockIdx.x * blockDim.x + threadIdx.x;
    if (i < BATCH) atomicAdd(&counts[src[i] * NCH + tgt[i]], 1);
}

__global__ void k_scan(const int* __restrict__ counts, int* __restrict__ offsets,
                       int* __restrict__ cursors, int* __restrict__ ntiles,
                       int* __restrict__ tile_bucket) {
    __shared__ int sh[NPAIR];
    int p = threadIdx.x;
    int tcnt = (counts[p] + ROWS - 1) / ROWS;   // tiles in this bucket
    sh[p] = tcnt;
    __syncthreads();
    if (p == 0) {
        int run = 0;
        for (int q = 0; q < NPAIR; ++q) { int c = sh[q]; sh[q] = run; run += c; }
        ntiles[0] = run;
    }
    __syncthreads();
    int tof = sh[p];
    offsets[p] = tof * ROWS;
    cursors[p] = tof * ROWS;
    for (int i = 0; i < tcnt; ++i) tile_bucket[tof + i] = p;
}

__global__ void k_scatter(const int* __restrict__ src, const int* __restrict__ tgt,
                          int* __restrict__ cursors, int* __restrict__ sorted) {
    int i = blockIdx.x * blockDim.x + threadIdx.x;
    if (i < BATCH) {
        int pos = atomicAdd(&cursors[src[i] * NCH + tgt[i]], 1);
        sorted[pos] = i;
    }
}

// ---------------------------------------------------------------------------
// One 32x128 -> 32x128 layer: out = in @ W^T + bias, via bf16-split WMMA.
// Two 16-row M-subtiles share every B fragment (halves weight traffic).
// inbuf/outbuf: LDS tiles [32][132] f32. whi/wlo: bf16 [128 out][128 in].
// ---------------------------------------------------------------------------
__device__ __forceinline__ void layer_32x128(
    const float* __restrict__ inbuf, float* __restrict__ outbuf,
    const unsigned short* __restrict__ whi, const unsigned short* __restrict__ wlo,
    const float* __restrict__ bias, int lane)
{
    const int g   = lane >> 4;    // half-wave: selects K sub-range per ISA layout
    const int n16 = lane & 15;    // A: row M (within subtile); B/C/D: column N

    v8f acc[2][8];
    #pragma unroll
    for (int mt = 0; mt < 2; ++mt)
        #pragma unroll
        for (int nt = 0; nt < 8; ++nt)
            #pragma unroll
            for (int q = 0; q < 8; ++q) acc[mt][nt][q] = 0.0f;

    #pragma unroll
    for (int kt = 0; kt < 4; ++kt) {                  // K = kt*32 .. +31
        // ---- build A fragments (hi/lo) for both M-subtiles (ISA 16-bit A layout)
        FragB16 ahi[2], alo[2];
        #pragma unroll
        for (int mt = 0; mt < 2; ++mt) {
            #pragma unroll
            for (int j = 0; j < 8; ++j) {
                int k0 = kt * 32 + ((j < 4) ? 2 * j : 16 + 2 * (j - 4)) + 8 * g;
                float f0 = inbuf[(mt * 16 + n16) * 132 + k0];
                float f1 = inbuf[(mt * 16 + n16) * 132 + k0 + 1];
                unsigned short h0, l0, h1, l1;
                split_bf16(f0, h0, l0);
                split_bf16(f1, h1, l1);
                ahi[mt].u[j] = (unsigned int)h0 | ((unsigned int)h1 << 16);
                alo[mt].u[j] = (unsigned int)l0 | ((unsigned int)l1 << 16);
            }
        }
        #pragma unroll
        for (int nt = 0; nt < 8; ++nt) {              // output cols nt*16 .. +15
            // ---- B fragments: B[k][n] = W[nt*16+n][k], per ISA 16-bit B layout
            const unsigned short* wr_hi = whi + (nt * 16 + n16) * DIM;
            const unsigned short* wr_lo = wlo + (nt * 16 + n16) * DIM;
            FragB16 bhi, blo;
            #pragma unroll
            for (int j = 0; j < 8; ++j) {
                int kb = kt * 32 + ((j < 4) ? 2 * j : 16 + 2 * (j - 4)) + 8 * g;    // even -> dword aligned
                bhi.u[j] = *(const unsigned int*)(wr_hi + kb);
                blo.u[j] = *(const unsigned int*)(wr_lo + kb);
            }
            #pragma unroll
            for (int mt = 0; mt < 2; ++mt) {
                // x*y ~= hi*hi + hi*lo + lo*hi  (small terms first)
                acc[mt][nt] = __builtin_amdgcn_wmma_f32_16x16x32_bf16(
                    false, alo[mt].bf, false, bhi.bf, (short)0, acc[mt][nt], false, false);
                acc[mt][nt] = __builtin_amdgcn_wmma_f32_16x16x32_bf16(
                    false, ahi[mt].bf, false, blo.bf, (short)0, acc[mt][nt], false, false);
                acc[mt][nt] = __builtin_amdgcn_wmma_f32_16x16x32_bf16(
                    false, ahi[mt].bf, false, bhi.bf, (short)0, acc[mt][nt], false, false);
            }
        }
    }

    // ---- bias + spill to LDS (C/D layout: lane holds col N, VGPR v -> row v+8g)
    #pragma unroll
    for (int nt = 0; nt < 8; ++nt) {
        float b = bias[nt * 16 + n16];
        #pragma unroll
        for (int mt = 0; mt < 2; ++mt)
            #pragma unroll
            for (int v = 0; v < 8; ++v)
                outbuf[(mt * 16 + v + 8 * g) * 132 + nt * 16 + n16] = acc[mt][nt][v] + b;
    }
}

// ---------------------------------------------------------------------------
__global__ __launch_bounds__(32) void k_jump(
    const float* __restrict__ z, const int* __restrict__ sorted,
    const int* __restrict__ tile_bucket, const int* __restrict__ ntiles,
    const unsigned short* __restrict__ ehi, const unsigned short* __restrict__ elo,
    const unsigned short* __restrict__ dhi, const unsigned short* __restrict__ dlo,
    const float* __restrict__ c_param, const float* __restrict__ d_param,
    float* __restrict__ out)
{
    int t = blockIdx.x;
    if (t >= ntiles[0]) return;
    int lane = threadIdx.x;

    __shared__ float abuf[ROWS * 132];   // activations (z, then final output)
    __shared__ float hbuf[ROWS * 132];   // hidden h
    __shared__ int   rid[ROWS];

    int bucket = tile_bucket[t];
    int src = bucket >> 4, tgt = bucket & 15;

    rid[lane] = sorted[t * ROWS + lane];
    __syncthreads();

    // ---- stage 32 gathered z rows into LDS (coalesced float4, pad rows = 0)
    #pragma unroll
    for (int it = 0; it < ROWS; ++it) {
        int f = it * 32 + lane;          // 1024 float4 chunks = 32 rows * 32
        int m = f >> 5, c4 = f & 31;
        int row = rid[m];
        v4f v = {0.f, 0.f, 0.f, 0.f};
        if (row >= 0) v = *(const v4f*)(z + row * DIM + c4 * 4);
        *(v4f*)(&abuf[m * 132 + c4 * 4]) = v;
    }
    __syncthreads();

    // encode: h = z @ enc_w[src]^T + c[src]
    layer_32x128(abuf, hbuf,
                 ehi + src * DIM * DIM, elo + src * DIM * DIM,
                 c_param + src * DIM, lane);
    __syncthreads();

    // decode: z_out = h @ dec_w[tgt]^T + d[tgt]   (reuse abuf as output tile)
    layer_32x128(hbuf, abuf,
                 dhi + tgt * DIM * DIM, dlo + tgt * DIM * DIM,
                 d_param + tgt * DIM, lane);
    __syncthreads();

    // ---- scatter result rows back (coalesced float4 within a row)
    #pragma unroll
    for (int it = 0; it < ROWS; ++it) {
        int f = it * 32 + lane;
        int m = f >> 5, c4 = f & 31;
        int row = rid[m];
        if (row >= 0) {
            v4f v = *(const v4f*)(&abuf[m * 132 + c4 * 4]);
            *(v4f*)(out + row * DIM + c4 * 4) = v;
        }
    }
}

// ---------------------------------------------------------------------------
extern "C" void kernel_launch(void* const* d_in, const int* in_sizes, int n_in,
                              void* d_out, int out_size, void* d_ws, size_t ws_size,
                              hipStream_t stream) {
    const float* z_n     = (const float*)d_in[0];
    const int*   src_idx = (const int*)d_in[1];
    const int*   tgt_idx = (const int*)d_in[2];
    const float* enc_w   = (const float*)d_in[3];
    const float* dec_w   = (const float*)d_in[4];
    const float* c_param = (const float*)d_in[5];
    const float* d_param = (const float*)d_in[6];
    float* out = (float*)d_out;

    int* W = (int*)d_ws;                       // needs ~2.7 MB of workspace
    int* counts  = W + WS_COUNTS;
    int* offsets = W + WS_OFFSETS;
    int* cursors = W + WS_CURSORS;
    int* ntiles  = W + WS_NTILES;
    int* tileb   = W + WS_TILEB;
    int* sorted  = W + WS_SORTED;
    unsigned short* ehi = (unsigned short*)(W + WS_WEIGHTS);
    unsigned short* elo = ehi + WELEMS;
    unsigned short* dhi = elo + WELEMS;
    unsigned short* dlo = dhi + WELEMS;
    (void)in_sizes; (void)n_in; (void)out_size; (void)ws_size; (void)offsets;

    k_init<<<(WELEMS + 255) / 256, 256, 0, stream>>>(enc_w, dec_w, counts, sorted,
                                                     ehi, elo, dhi, dlo);
    k_count<<<BATCH / 256, 256, 0, stream>>>(src_idx, tgt_idx, counts);
    k_scan<<<1, NPAIR, 0, stream>>>(counts, offsets, cursors, ntiles, tileb);
    k_scatter<<<BATCH / 256, 256, 0, stream>>>(src_idx, tgt_idx, cursors, sorted);
    k_jump<<<NTILE_MAX, 32, 0, stream>>>(z_n, sorted, tileb, ntiles,
                                         ehi, elo, dhi, dlo,
                                         c_param, d_param, out);
}
